// NTM_29059748724813
// MI455X (gfx1250) — matrix operand order repaired and checked by reading, not compiled, not run on hardware
//
#include <hip/hip_runtime.h>
#include <hip/hip_bf16.h>
#include <stdint.h>

#define DEVI __device__ __forceinline__

typedef __attribute__((ext_vector_type(16))) __bf16 v16bf;
typedef __attribute__((ext_vector_type(8)))  __bf16 v8bf;
typedef __attribute__((ext_vector_type(8)))  float  v8f;
typedef int v4i_ __attribute__((vector_size(16)));   // matches builtin prototype

#if __has_builtin(__builtin_amdgcn_global_load_async_to_lds_b128) && \
    __has_builtin(__builtin_amdgcn_s_wait_asynccnt)
#define HAVE_ASYNC_LDS 1
#else
#define HAVE_ASYNC_LDS 0
#endif

// ---- problem sizes ----
constexpr int B_ = 32, S_ = 256, I_ = 256, H_ = 512, N_ = 128, C_ = 64, O_ = 256, P_ = 268;
constexpr int ZD   = I_ + C_ + H_;   // 832  z = [x(256) | r(64) | h(512)]
constexpr int NWG  = 16;             // persistent workgroups
constexpr int TPB  = 512;            // 16 wave32
constexpr int KST_L = ZD / 32;       // 26 k-steps, LSTM GEMM
constexpr int NT_H  = 17;            // head N tiles (268 -> 272)
constexpr int KST_H = H_ / 32;       // 16
constexpr int NT_O  = O_ / 16;       // 16
constexpr int KST_O = (H_ + C_) / 32; // 18
constexpr float EPSF = 1e-8f;

// ---- workspace layout (bytes) ----
constexpr size_t OFF_WCAT = 0;                                  // bf16 [16][8][26][512]
constexpr size_t SZ_WCAT  = (size_t)NWG * 8 * KST_L * 512 * 2;
constexpr size_t OFF_WHEAD = OFF_WCAT + SZ_WCAT;                // bf16 [17][16][512]
constexpr size_t SZ_WHEAD  = (size_t)NT_H * KST_H * 512 * 2;
constexpr size_t OFF_WOUT  = OFF_WHEAD + SZ_WHEAD;              // bf16 [16][18][512]
constexpr size_t SZ_WOUT   = (size_t)NT_O * KST_O * 512 * 2;
constexpr size_t OFF_Z0    = OFF_WOUT + SZ_WOUT;                // bf16 [32][832]
constexpr size_t SZ_Z      = (size_t)B_ * ZD * 2;
constexpr size_t OFF_Z1    = OFF_Z0 + SZ_Z;
constexpr size_t OFF_C     = OFF_Z1 + SZ_Z;                     // f32 [32][512]
constexpr size_t OFF_M     = OFF_C  + (size_t)B_ * H_ * 4;      // f32 [32][128][64]
constexpr size_t OFF_WR    = OFF_M  + (size_t)B_ * N_ * C_ * 4; // f32 [32][128]
constexpr size_t OFF_WW    = OFF_WR + (size_t)B_ * N_ * 4;
constexpr size_t OFF_P     = OFF_WW + (size_t)B_ * N_ * 4;      // f32 [32][268]
constexpr size_t OFF_BAR   = (OFF_P + (size_t)B_ * P_ * 4 + 255) & ~(size_t)255;

DEVI float softplusf(float x) { return (x > 20.f) ? x : log1pf(__expf(x)); }
DEVI float sigmoidf(float x)  { return 1.f / (1.f + __expf(-x)); }

DEVI v8f wmma_bf16(v16bf a, v16bf b, v8f c) {
  return __builtin_amdgcn_wmma_f32_16x16x32_bf16(false, a, false, b, (short)0, c, false, false);
}

// A fragment from LDS z tile: lanes 0-15 -> M=0..15 (khalf 0), lanes 16-31 -> khalf 1.
// Per ISA 16-bit A layout, each lane holds k = kb..kb+7 and kb+16..kb+23 (two b128 LDS loads).
DEVI v16bf loadA(const __bf16* lz, int mtile, int lane, int zlo, int zhi) {
  int m = mtile * 16 + (lane & 15);
  const v8bf lo = *reinterpret_cast<const v8bf*>(lz + m * ZD + zlo);
  const v8bf hi = *reinterpret_cast<const v8bf*>(lz + m * ZD + zhi);
  v16bf a;
#pragma unroll
  for (int e = 0; e < 8; ++e) { a[e] = lo[e]; a[8 + e] = hi[e]; }
  return a;
}

// B fragment: fragment-linear packed weights, 32 contiguous bytes per lane.
DEVI v16bf loadB(const __bf16* wp, int lane) {
  return *reinterpret_cast<const v16bf*>(wp + lane * 16);
}

// Stage 53KB z vector (bf16 [32][832]) from L2 into LDS. Prefer the gfx1250
// async LDS-DMA path (no VGPR round-trip, tracked by ASYNCcnt); fall back to
// a register copy when the builtin isn't declared by the toolchain.
DEVI void stage_z(const __bf16* zg, char* shp) {
#if HAVE_ASYNC_LDS
  for (int i = threadIdx.x; i < (B_ * ZD * 2) / 16; i += TPB) {
    __builtin_amdgcn_global_load_async_to_lds_b128(
        (v4i_*)((char*)zg + (size_t)i * 16),
        (v4i_*)(shp + (size_t)i * 16), 0, 0);
  }
  __builtin_amdgcn_s_wait_asynccnt(0);
#else
  const uint32_t* g = (const uint32_t*)zg;
  uint32_t* l = (uint32_t*)shp;
  for (int i = threadIdx.x; i < B_ * ZD / 2; i += TPB) l[i] = g[i];
#endif
}

DEVI void gsync(unsigned* bar) {
  __threadfence();
  __syncthreads();
  if (threadIdx.x == 0) {
    unsigned old = __hip_atomic_fetch_add(bar, 1u, __ATOMIC_ACQ_REL, __HIP_MEMORY_SCOPE_AGENT);
    unsigned target = (old / NWG + 1u) * NWG;
    while (__hip_atomic_load(bar, __ATOMIC_ACQUIRE, __HIP_MEMORY_SCOPE_AGENT) < target)
      __builtin_amdgcn_s_sleep(8);
  }
  __syncthreads();
}

// -------------------- init / repack --------------------
__global__ void ntm_init(const float* __restrict__ xin, const float* __restrict__ Wih,
                         const float* __restrict__ Whh, const float* __restrict__ Whd,
                         const float* __restrict__ Wou, char* __restrict__ ws) {
  __bf16* wc  = (__bf16*)(ws + OFF_WCAT);
  __bf16* whp = (__bf16*)(ws + OFF_WHEAD);
  __bf16* wop = (__bf16*)(ws + OFF_WOUT);
  __bf16* z0  = (__bf16*)(ws + OFF_Z0);
  float*  Cst = (float*)(ws + OFF_C);
  float*  Mm  = (float*)(ws + OFF_M);
  float*  Wr  = (float*)(ws + OFF_WR);
  float*  Ww  = (float*)(ws + OFF_WW);
  unsigned* bar = (unsigned*)(ws + OFF_BAR);

  size_t tid = (size_t)blockIdx.x * blockDim.x + threadIdx.x;
  size_t nth = (size_t)gridDim.x * blockDim.x;

  // LSTM weights: gate-interleaved per WG, fragment-linear B tiles.
  for (size_t d = tid; d < (size_t)NWG * 8 * KST_L * 512; d += nth) {
    int i = (int)(d & 511); int blk = (int)(d >> 9);
    int kk = blk % KST_L; int t = (blk / KST_L) & 7; int wg = blk / (KST_L * 8);
    int lane = i >> 4, e = i & 15;
    int n = lane & 15;
    int k = kk * 32 + (lane >> 4) * 8 + (e & 7) + (e >> 3) * 16; // z index 0..831
    int col = (t >> 1) * H_ + wg * 32 + (t & 1) * 16 + n; // gate-major row in [4H]
    float v = (k < I_ + C_) ? Wih[(size_t)col * (I_ + C_) + k]
                            : Whh[(size_t)col * H_ + (k - (I_ + C_))];
    wc[d] = (__bf16)v;
  }
  // Head weights
  for (size_t d = tid; d < (size_t)NT_H * KST_H * 512; d += nth) {
    int i = (int)(d & 511); int blk = (int)(d >> 9);
    int kk = blk & 15; int tile = blk >> 4;
    int lane = i >> 4, e = i & 15;
    int n = lane & 15;
    int k = kk * 32 + (lane >> 4) * 8 + (e & 7) + (e >> 3) * 16;
    int col = tile * 16 + n;
    whp[d] = (__bf16)((col < P_) ? Whd[(size_t)col * H_ + k] : 0.f);
  }
  // Output weights
  for (size_t d = tid; d < (size_t)NT_O * KST_O * 512; d += nth) {
    int i = (int)(d & 511); int blk = (int)(d >> 9);
    int kk = blk % KST_O; int tile = blk / KST_O;
    int lane = i >> 4, e = i & 15;
    int n = lane & 15;
    int k = kk * 32 + (lane >> 4) * 8 + (e & 7) + (e >> 3) * 16; // [h(512)|r(64)]
    int col = tile * 16 + n;
    wop[d] = (__bf16)Wou[(size_t)col * (H_ + C_) + k];
  }
  // z0 = [x_0 | 0 | 0]
  for (size_t d = tid; d < (size_t)B_ * ZD; d += nth) {
    int b = (int)(d / ZD); int k = (int)(d % ZD);
    float v = (k < I_) ? xin[(size_t)b * S_ * I_ + k] : 0.f;
    z0[d] = (__bf16)v;
  }
  for (size_t d = tid; d < (size_t)B_ * H_; d += nth)       Cst[d] = 0.f;
  for (size_t d = tid; d < (size_t)B_ * N_ * C_; d += nth)  Mm[d] = 0.01f;
  for (size_t d = tid; d < (size_t)B_ * N_; d += nth) {
    float v = ((d % N_) == 0) ? 1.f : 0.f;
    Wr[d] = v; Ww[d] = v;
  }
  if (tid == 0) *bar = 0u;
}

// -------------------- persistent NTM kernel --------------------
__global__ __launch_bounds__(TPB, 1)
void ntm_persist(const float* __restrict__ xin, const float* __restrict__ bl,
                 const float* __restrict__ bh, const float* __restrict__ bo,
                 float* __restrict__ out, char* __restrict__ ws) {
  __shared__ __align__(16) char sh[56 * 1024];
  __bf16* lz   = (__bf16*)sh;                     // [32][832] bf16 z staging
  float*  gbuf = (float*)sh;                      // [32][4][32] gates (aliased after sync)

  const int wg   = blockIdx.x;
  const int wv   = threadIdx.x >> 5;
  const int lane = threadIdx.x & 31;

  const __bf16* WC = (const __bf16*)(ws + OFF_WCAT) + (size_t)wg * 8 * KST_L * 512;
  const __bf16* WH = (const __bf16*)(ws + OFF_WHEAD);
  const __bf16* WO = (const __bf16*)(ws + OFF_WOUT);
  __bf16* zbuf[2] = { (__bf16*)(ws + OFF_Z0), (__bf16*)(ws + OFF_Z1) };
  float* Cst = (float*)(ws + OFF_C);
  float* Mm  = (float*)(ws + OFF_M);
  float* Wr  = (float*)(ws + OFF_WR);
  float* Ww  = (float*)(ws + OFF_WW);
  float* Pp  = (float*)(ws + OFF_P);
  unsigned* bar = (unsigned*)(ws + OFF_BAR);

  for (int t = 0; t < S_; ++t) {
    __bf16* zc = zbuf[t & 1];
    __bf16* zn = zbuf[(t + 1) & 1];

    // ---- stage z_cur (x_t, r_{t-1}, h_{t-1}) into LDS ----
    stage_z(zc, sh);
    __syncthreads();

    // ---- LSTM GEMM: this WG computes 4 gates x 32 hidden for all 32 batch rows ----
    const int mt = wv & 1;       // batch tile
    const int st = wv >> 1;      // 8 N-subtiles: gate = st>>1, half = st&1
    v8f acc = {0.f, 0.f, 0.f, 0.f, 0.f, 0.f, 0.f, 0.f};
    {
      const __bf16* wblk = WC + (size_t)st * KST_L * 512;
      for (int kk = 0; kk < KST_L; ++kk) {
        __builtin_prefetch(wblk + (size_t)(kk + 1) * 512, 0, 1);
        int kb = kk * 32 + (lane >> 4) * 8;
        v16bf a = loadA(lz, mt, lane, kb, kb + 16);
        v16bf bf = loadB(wblk + (size_t)kk * 512, lane);
        acc = wmma_bf16(a, bf, acc);
      }
    }
    __syncthreads();  // all waves done reading lz; safe to alias as gbuf

    // ---- scatter gate tiles (+bias) to LDS ----
    {
      int gate = st >> 1;
      int jl = (st & 1) * 16 + (lane & 15);
      float bia = bl[gate * H_ + wg * 32 + jl];
#pragma unroll
      for (int v = 0; v < 8; ++v) {
        int bidx = mt * 16 + v + (lane >> 4) * 8;
        gbuf[(bidx * 4 + gate) * 32 + jl] = acc[v] + bia;
      }
    }
    __syncthreads();

    // ---- LSTM nonlinearity for this WG's 32 hidden units ----
    for (int i = threadIdx.x; i < B_ * 32; i += TPB) {
      int b = i >> 5, jl = i & 31;
      float gi = gbuf[(b * 4 + 0) * 32 + jl];
      float gf = gbuf[(b * 4 + 1) * 32 + jl];
      float gg = gbuf[(b * 4 + 2) * 32 + jl];
      float go = gbuf[(b * 4 + 3) * 32 + jl];
      int j = wg * 32 + jl;
      float cc = Cst[b * H_ + j];
      float cn = sigmoidf(gf) * cc + sigmoidf(gi) * tanhf(gg);
      Cst[b * H_ + j] = cn;
      zn[b * ZD + I_ + C_ + j] = (__bf16)(sigmoidf(go) * tanhf(cn));
    }
    gsync(bar);  // h_t visible everywhere

    // ---- head GEMM: p = h @ W_head^T ----
    stage_z(zn, sh);
    __syncthreads();
    {
      int tile = -1, mt2 = 0;
      if (wv < 2) { tile = wg; mt2 = wv; }
      else if (wg == 0 && wv < 4) { tile = 16; mt2 = wv - 2; }
      if (tile >= 0) {
        v8f a2 = {0.f, 0.f, 0.f, 0.f, 0.f, 0.f, 0.f, 0.f};
        const __bf16* wb = WH + (size_t)tile * KST_H * 512;
#pragma unroll 4
        for (int kk = 0; kk < KST_H; ++kk) {
          int kb = kk * 32 + (lane >> 4) * 8;
          v16bf a = loadA(lz, mt2, lane, 320 + kb, 320 + kb + 16);
          v16bf bf = loadB(wb + (size_t)kk * 512, lane);
          a2 = wmma_bf16(a, bf, a2);
        }
        int col = tile * 16 + (lane & 15);
        if (col < P_) {
          float bia = bh[col];
#pragma unroll
          for (int v = 0; v < 8; ++v) {
            int b = mt2 * 16 + v + (lane >> 4) * 8;
            Pp[b * P_ + col] = a2[v] + bia;
          }
        }
      }
    }
    gsync(bar);  // p visible

    // ---- NTM addressing + memory update + read (2 batches per WG, 4x128 threads) ----
    {
      float* sP  = (float*)sh;          // [4][272]
      float* sW  = sP + 4 * 272;        // [4][128]
      float* sRd = sW + 4 * 128;        // [4][128]
      int grp = threadIdx.x >> 7;       // (batch,head) group
      int tn  = threadIdx.x & 127;      // address index n
      int b    = wg * 2 + (grp >> 1);
      int head = grp & 1;               // 0=read, 1=write
      float* myP = sP + grp * 272;
      float* myW = sW + grp * 128;
      float* myR = sRd + grp * 128;

      for (int i = tn; i < P_; i += 128) myP[i] = Pp[b * P_ + i];
      __syncthreads();

      int base = head ? 70 : 0;
      float bta = softplusf(myP[base + 64]);
      float gte = sigmoidf(myP[base + 65]);
      float s0 = myP[base + 66], s1 = myP[base + 67], s2 = myP[base + 68];
      float smx = fmaxf(s0, fmaxf(s1, s2));
      float e0 = __expf(s0 - smx), e1 = __expf(s1 - smx), e2 = __expf(s2 - smx);
      float esm = e0 + e1 + e2; e0 /= esm; e1 /= esm; e2 /= esm;
      float gma = 1.f + softplusf(myP[base + 69]);

      float kn2 = 0.f;
#pragma unroll 8
      for (int c = 0; c < C_; ++c) { float kv = myP[base + c]; kn2 += kv * kv; }
      float kinv = 1.f / (sqrtf(kn2) + EPSF);

      const float* Mrow = Mm + ((size_t)b * N_ + tn) * C_;
      float dot = 0.f, mn2 = 0.f;
#pragma unroll 8
      for (int c = 0; c < C_; ++c) { float mv = Mrow[c]; dot += mv * myP[base + c]; mn2 += mv * mv; }
      float sc = bta * (dot * kinv / (sqrtf(mn2) + EPSF));

      myR[tn] = sc; __syncthreads();
      for (int off = 64; off; off >>= 1) { if (tn < off) myR[tn] = fmaxf(myR[tn], myR[tn + off]); __syncthreads(); }
      float mx = myR[0]; __syncthreads();
      float ex = __expf(sc - mx);
      myR[tn] = ex; __syncthreads();
      for (int off = 64; off; off >>= 1) { if (tn < off) myR[tn] += myR[tn + off]; __syncthreads(); }
      float wcn = ex / myR[0]; __syncthreads();

      float* wprev = head ? Ww : Wr;
      float wgv = gte * wcn + (1.f - gte) * wprev[b * N_ + tn];
      myW[tn] = wgv; __syncthreads();
      float wsv = e0 * myW[(tn + 1) & 127] + e1 * wgv + e2 * myW[(tn + 127) & 127];
      float wpv = __powf(wsv + EPSF, gma);
      myR[tn] = wpv; __syncthreads();
      for (int off = 64; off; off >>= 1) { if (tn < off) myR[tn] += myR[tn + off]; __syncthreads(); }
      float wnew = wpv / myR[0]; __syncthreads();
      myW[tn] = wnew;
      wprev[b * N_ + tn] = wnew;
      __syncthreads();

      if (head == 1) {  // erase + add
        float wwn = myW[tn];
        float* Mr = Mm + ((size_t)b * N_ + tn) * C_;
#pragma unroll 4
        for (int c = 0; c < C_; ++c) {
          float er = sigmoidf(myP[140 + c]);
          Mr[c] = Mr[c] * (1.f - wwn * er) + wwn * myP[204 + c];
        }
      }
      __syncthreads();
      if (head == 0 && tn < C_) {  // read with new M
        const float* Mb = Mm + (size_t)b * N_ * C_;
        float rc = 0.f;
        for (int n = 0; n < N_; ++n) rc += myW[n] * Mb[n * C_ + tn];
        zn[b * ZD + I_ + tn] = (__bf16)rc;
      }
    }
    gsync(bar);  // r_t, M_t visible

    // ---- output GEMM: out = [h,r] @ W_out^T ----
    stage_z(zn, sh);
    __syncthreads();
    if (wv < 2) {
      v8f a3 = {0.f, 0.f, 0.f, 0.f, 0.f, 0.f, 0.f, 0.f};
      const __bf16* wb = WO + (size_t)wg * KST_O * 512;
#pragma unroll 3
      for (int kk = 0; kk < KST_O; ++kk) {
        int kb = kk * 32 + (lane >> 4) * 8;
        int zlo = (kb < 512) ? 320 + kb : kb - 256;        // h at z+320, r at z+256
        int kh = kb + 16;
        int zhi = (kh < 512) ? 320 + kh : kh - 256;
        v16bf a = loadA(lz, wv, lane, zlo, zhi);
        v16bf bf = loadB(wb + (size_t)kk * 512, lane);
        a3 = wmma_bf16(a, bf, a3);
      }
      int col = wg * 16 + (lane & 15);
      float bia = bo[col];
#pragma unroll
      for (int v = 0; v < 8; ++v) {
        int b = wv * 16 + v + (lane >> 4) * 8;
        out[((size_t)b * S_ + t) * O_ + col] = a3[v] + bia;
      }
    }

    // ---- load x_{t+1} into z_next ----
    if (t + 1 < S_) {
      for (int i = wg * TPB + threadIdx.x; i < B_ * I_; i += NWG * TPB) {
        int b = i >> 8, k = i & 255;
        zn[b * ZD + k] = (__bf16)xin[((size_t)b * S_ + (t + 1)) * I_ + k];
      }
    }
    gsync(bar);  // end of step
  }
}

extern "C" void kernel_launch(void* const* d_in, const int* in_sizes, int n_in,
                              void* d_out, int out_size, void* d_ws, size_t ws_size,
                              hipStream_t stream) {
  (void)in_sizes; (void)n_in; (void)out_size; (void)ws_size;
  const float* xin = (const float*)d_in[0];
  const float* Wih = (const float*)d_in[1];
  const float* Whh = (const float*)d_in[2];
  const float* bl  = (const float*)d_in[3];
  const float* Whd = (const float*)d_in[4];
  const float* bh  = (const float*)d_in[5];
  const float* Wou = (const float*)d_in[6];
  const float* bo  = (const float*)d_in[7];
  float* out = (float*)d_out;
  char* ws = (char*)d_ws;

  ntm_init<<<512, 256, 0, stream>>>(xin, Wih, Whh, Whd, Wou, ws);
  ntm_persist<<<NWG, TPB, 0, stream>>>(xin, bl, bh, bo, out, ws);
}
